// SelfAttnBlock_4037269258665
// MI455X (gfx1250) — compile-verified, matrix operand order checked by reading, not run on hardware
//
#include <hip/hip_runtime.h>

// SAGAN self-attention block for MI455X (gfx1250, wave32, WMMA).
// bf16 WMMA everywhere; flash-attention online softmax computed as
// S^T = K*Q^T so probabilities land directly in B-operand layout for
// O^T = V^T * P^T (no register transposes); V stored transposed at projection.
// K/V^T tiles staged into LDS with gfx1250 async-to-LDS copies (ASYNCcnt).

typedef __attribute__((ext_vector_type(16))) __bf16 v16bf;
typedef __attribute__((ext_vector_type(8)))  __bf16 v8bf;
typedef __attribute__((ext_vector_type(8)))  float  v8f;

#define BATCH 4
#define HW    4096      // 64*64 tokens per batch
#define CCH   256       // channels
#define DD    32        // q/k head dim = C/8
#define MTOT  (BATCH*HW)

// Load a 16x32 bf16 WMMA operand fragment (A or B — layouts are symmetric):
// lane<16 holds row (row0+lane), K-chunks [koff, koff+8) and [koff+16, koff+24);
// lane>=16 holds same row with chunks shifted by +8.
__device__ __forceinline__ v16bf load_frag(const __bf16* __restrict__ base,
                                           int row0, int ld, int koff, int lane) {
  const int lm = lane & 15;
  const int h  = lane >> 4;
  const __bf16* p = base + (size_t)(row0 + lm) * ld + koff + h * 8;
  v8bf c0 = *(const v8bf*)(p);
  v8bf c1 = *(const v8bf*)(p + 16);
  v16bf r;
#pragma unroll
  for (int i = 0; i < 8; ++i) { r[i] = c0[i]; r[i + 8] = c1[i]; }
  return r;
}

// gfx1250 async global->LDS 16B copy (tracked by ASYNCcnt, no VGPR data path).
__device__ __forceinline__ void async_copy_b128(void* lds_dst, const void* gsrc) {
  unsigned lds_addr = (unsigned)(unsigned long long)lds_dst;   // low 32b = LDS offset
  unsigned long long gaddr = (unsigned long long)gsrc;
  asm volatile("global_load_async_to_lds_b128 %0, %1, off"
               :
               : "v"(lds_addr), "v"(gaddr)
               : "memory");
}

__device__ __forceinline__ void wait_async_zero() {
  asm volatile("s_wait_asynccnt 0x0" ::: "memory");
}

// ---------------- prep kernels ----------------

__global__ void k_cvt_bf16(const float* __restrict__ src, __bf16* __restrict__ dst, int n) {
  int i = blockIdx.x * blockDim.x + threadIdx.x;
  if (i < n) dst[i] = (__bf16)src[i];
}

// W: rows x cols (row-major f32) -> WT: cols x rows (row-major bf16)
__global__ void k_transpose_bf16(const float* __restrict__ W, __bf16* __restrict__ WT,
                                 int rows, int cols) {
  int i = blockIdx.x * blockDim.x + threadIdx.x;
  if (i < rows * cols) {
    int r = i / cols, c = i % cols;
    WT[(size_t)c * rows + r] = (__bf16)W[i];
  }
}

// ---------------- fused QKV projection ----------------
// Each wave: 16 rows of x, all 320 output columns (20 tiles: q[2] k[2] v[16]).
// q,k stored row-major bf16 ((B*HW) x 32); v stored transposed: vbT[b][c][pix].
__global__ void __launch_bounds__(128)
k_proj(const __bf16* __restrict__ xb,
       const __bf16* __restrict__ WqT, const __bf16* __restrict__ WkT,
       const __bf16* __restrict__ WvT,
       const float* __restrict__ bq, const float* __restrict__ bk,
       const float* __restrict__ bv,
       __bf16* __restrict__ qb, __bf16* __restrict__ kb, __bf16* __restrict__ vbT) {
  const int lane = threadIdx.x & 31;
  const int wave = threadIdx.x >> 5;
  const int row0 = blockIdx.x * 64 + wave * 16;
  const int lm = lane & 15, h = lane >> 4;

  v8f acc[20];
#pragma unroll
  for (int t = 0; t < 20; ++t)
#pragma unroll
    for (int v = 0; v < 8; ++v) acc[t][v] = 0.f;

  for (int k0 = 0; k0 < CCH; k0 += 32) {
    v16bf a = load_frag(xb, row0, CCH, k0, lane);
#pragma unroll
    for (int t = 0; t < 20; ++t) {
      const __bf16* wt = (t < 2) ? (WqT + (size_t)t * 16 * CCH)
                       : (t < 4) ? (WkT + (size_t)(t - 2) * 16 * CCH)
                                 : (WvT + (size_t)(t - 4) * 16 * CCH);
      v16bf bfr = load_frag(wt, 0, CCH, k0, lane);
      acc[t] = __builtin_amdgcn_wmma_f32_16x16x32_bf16(
          false, a, false, bfr, (short)0, acc[t], false, false);
    }
  }

  const int b    = row0 / HW;
  const int pix0 = row0 % HW;
#pragma unroll
  for (int t = 0; t < 20; ++t) {
    if (t < 4) {
      const float* bias = (t < 2) ? bq : bk;
      __bf16* dst = (t < 2) ? qb : kb;
      int col = ((t & 1) ? 16 : 0) + lm;
      float bb = bias[col];
#pragma unroll
      for (int v = 0; v < 8; ++v) {
        int r = row0 + v + 8 * h;
        dst[(size_t)r * DD + col] = (__bf16)(acc[t][v] + bb);
      }
    } else {
      int c = (t - 4) * 16 + lm;
      float bb = bv[c];
      v8bf pv;
#pragma unroll
      for (int v = 0; v < 8; ++v) pv[v] = (__bf16)(acc[t][v] + bb);
      *(v8bf*)(vbT + (size_t)b * (CCH * HW) + (size_t)c * HW + pix0 + 8 * h) = pv;
    }
  }
}

// ---------------- flash attention ----------------
// Block: 128 threads = 4 waves, each wave owns 16 query rows.
// Loop over 64-key blocks; K (64x32) and V^T (256x64) staged in LDS via
// async-to-LDS copies. S^T = K*Q^T -> lane owns one query column -> softmax
// via shfl_xor(16). O^T += V^T * P^T with P frags built in-lane.
__global__ void __launch_bounds__(128)
k_attn(const __bf16* __restrict__ qb, const __bf16* __restrict__ kb,
       const __bf16* __restrict__ vbT,
       const float* __restrict__ x, const float* __restrict__ gamma_p,
       float* __restrict__ out) {
  __shared__ __bf16 Klds[64 * 40];      // padded row stride 40
  __shared__ __bf16 Vtlds[256 * 72];    // padded row stride 72

  const int tid  = threadIdx.x;
  const int lane = tid & 31;
  const int wave = tid >> 5;
  const int lm = lane & 15, h = lane >> 4;
  const int b    = blockIdx.y;
  const int q0w  = blockIdx.x * 64 + wave * 16;
  const float gamma = gamma_p[0];

  // Q^T operand (32 x 16): one frag per wave, reused across all key blocks.
  const __bf16* qbase = qb + (size_t)(b * HW + q0w) * DD;
  v16bf qt = load_frag(qbase, 0, DD, 0, lane);

  v8f o[16];
#pragma unroll
  for (int t = 0; t < 16; ++t)
#pragma unroll
    for (int v = 0; v < 8; ++v) o[t][v] = 0.f;

  float m = -1e30f, l = 0.f;

  const __bf16* vtb_base = vbT + (size_t)b * (CCH * HW);
  const __bf16* kb_base  = kb + (size_t)b * HW * DD;

  for (int j0 = 0; j0 < HW; j0 += 64) {
    // --- stage K block: 64 rows x 32 bf16, async global->LDS 16B copies ---
    for (int ch = tid; ch < 256; ch += 128) {
      int row = ch >> 2, off = (ch & 3) * 8;
      async_copy_b128(&Klds[row * 40 + off],
                      kb_base + (size_t)(j0 + row) * DD + off);
    }
    // --- stage V^T block: 256 rows x 64 keys, async global->LDS ---
    const __bf16* vtb = vtb_base + j0;
    for (int ch = tid; ch < 2048; ch += 128) {
      int row = ch >> 3, off = (ch & 7) * 8;
      async_copy_b128(&Vtlds[row * 72 + off],
                      vtb + (size_t)row * HW + off);
    }
    // prefetch next key block's V^T slice toward L2 while we compute
    if (j0 + 64 < HW) {
      __builtin_prefetch(vtb_base + (size_t)(tid * 2) * HW + (j0 + 64), 0, 1);
    }
    wait_async_zero();     // this wave's async copies landed in LDS
    __syncthreads();       // all waves' copies visible

    // --- S^T tiles: 4 x (16 keys x 16 queries), K-dim = D = 32 in one WMMA ---
    v8f s[4];
#pragma unroll
    for (int jt = 0; jt < 4; ++jt) {
      v16bf kf = load_frag(Klds, jt * 16, 40, 0, lane);
      v8f z;
#pragma unroll
      for (int v = 0; v < 8; ++v) z[v] = 0.f;
      s[jt] = __builtin_amdgcn_wmma_f32_16x16x32_bf16(
          false, kf, false, qt, (short)0, z, false, false);
    }

    // --- online softmax over this 64-key block (per query = per lane col) ---
    float mb = -1e30f;
#pragma unroll
    for (int jt = 0; jt < 4; ++jt)
#pragma unroll
      for (int v = 0; v < 8; ++v) mb = fmaxf(mb, s[jt][v]);
    mb = fmaxf(mb, __shfl_xor(mb, 16, 32));
    float mn    = fmaxf(m, mb);
    float scale = __expf(m - mn);
    float rs = 0.f;
    v8bf pb[4];
#pragma unroll
    for (int jt = 0; jt < 4; ++jt)
#pragma unroll
      for (int v = 0; v < 8; ++v) {
        float p = __expf(s[jt][v] - mn);
        rs += p;
        pb[jt][v] = (__bf16)p;
      }
    rs += __shfl_xor(rs, 16, 32);
    l = l * scale + rs;
    m = mn;
#pragma unroll
    for (int t = 0; t < 16; ++t)
#pragma unroll
      for (int v = 0; v < 8; ++v) o[t][v] *= scale;

    // --- P^T operand frags: pure in-lane repack (keys 0..31 / 32..63) ---
    v16bf pf[2];
#pragma unroll
    for (int kk = 0; kk < 2; ++kk)
#pragma unroll
      for (int i = 0; i < 8; ++i) {
        pf[kk][i]     = pb[2 * kk][i];
        pf[kk][i + 8] = pb[2 * kk + 1][i];
      }

    // --- O^T += V^T * P^T : 16 column tiles x 2 key chunks ---
#pragma unroll
    for (int ct = 0; ct < 16; ++ct)
#pragma unroll
      for (int kk = 0; kk < 2; ++kk) {
        v16bf vf = load_frag(Vtlds, ct * 16, 72, kk * 32, lane);
        o[ct] = __builtin_amdgcn_wmma_f32_16x16x32_bf16(
            false, vf, false, pf[kk], (short)0, o[ct], false, false);
      }
    __syncthreads();   // done reading LDS before next block's async writes
  }

  // --- epilogue: out[q][c] = gamma * O[q][c]/l + x[q][c] (vectorized 32B) ---
  float invl = 1.f / l;
  int q = q0w + lm;
  const float* xr = x + (size_t)(b * HW + q) * CCH;
  float* orow = out + (size_t)(b * HW + q) * CCH;
#pragma unroll
  for (int ct = 0; ct < 16; ++ct) {
    int c0 = ct * 16 + 8 * h;
    v8f xv = *(const v8f*)(xr + c0);
    v8f res;
#pragma unroll
    for (int v = 0; v < 8; ++v)
      res[v] = gamma * (o[ct][v] * invl) + xv[v];
    *(v8f*)(orow + c0) = res;
  }
}

// ---------------- launcher ----------------

extern "C" void kernel_launch(void* const* d_in, const int* in_sizes, int n_in,
                              void* d_out, int out_size, void* d_ws, size_t ws_size,
                              hipStream_t stream) {
  (void)in_sizes; (void)n_in; (void)out_size; (void)ws_size;
  const float* x     = (const float*)d_in[0];
  const float* Wq    = (const float*)d_in[1];
  const float* bq    = (const float*)d_in[2];
  const float* Wk    = (const float*)d_in[3];
  const float* bk    = (const float*)d_in[4];
  const float* Wv    = (const float*)d_in[5];
  const float* bv    = (const float*)d_in[6];
  const float* gamma = (const float*)d_in[7];
  float* out = (float*)d_out;

  char* ws = (char*)d_ws;
  __bf16* xb   = (__bf16*)(ws + 0);              // 16384*256*2 = 8,388,608
  __bf16* qbuf = (__bf16*)(ws + 8388608);        // 16384*32*2  = 1,048,576
  __bf16* kbuf = (__bf16*)(ws + 9437184);        // 1,048,576
  __bf16* vbT  = (__bf16*)(ws + 10485760);       // 8,388,608 (transposed V)
  __bf16* WqT  = (__bf16*)(ws + 18874368);       // 16,384
  __bf16* WkT  = (__bf16*)(ws + 18890752);       // 16,384
  __bf16* WvT  = (__bf16*)(ws + 18907136);       // 131,072

  const int NX = MTOT * CCH;  // 4,194,304
  k_cvt_bf16<<<(NX + 255) / 256, 256, 0, stream>>>(x, xb, NX);
  k_transpose_bf16<<<(CCH * DD + 255) / 256, 256, 0, stream>>>(Wq, WqT, CCH, DD);
  k_transpose_bf16<<<(CCH * DD + 255) / 256, 256, 0, stream>>>(Wk, WkT, CCH, DD);
  k_transpose_bf16<<<(CCH * CCH + 255) / 256, 256, 0, stream>>>(Wv, WvT, CCH, CCH);

  k_proj<<<MTOT / 64, 128, 0, stream>>>(xb, WqT, WkT, WvT, bq, bk, bv,
                                        qbuf, kbuf, vbT);

  k_attn<<<dim3(HW / 64, BATCH), 128, 0, stream>>>(qbuf, kbuf, vbT, x, gamma, out);
}
